// AnchAttention_69578470195682
// MI455X (gfx1250) — matrix operand order repaired and checked by reading, not compiled, not run on hardware
//
#include <hip/hip_runtime.h>
#include <hip/hip_bf16.h>

typedef __attribute__((ext_vector_type(16))) __bf16 v16bf;
typedef __attribute__((ext_vector_type(8)))  float  v8f;
typedef int vint4 __attribute__((vector_size(16)));   // matches builtin's V4i

static constexpr int H    = 512;
static constexpr int NVAR = 16384;
static constexpr int NP   = 4096;
static constexpr int NM   = 4096;
static constexpr float NEGV   = -1e30f;
static constexpr float RSQRTH = 0.04419417382415922f; // 1/sqrt(512)

// LDS tile geometry: A tile 128 rows x 520 bf16 (512 + 8 pad), B chunk 128 x 136 bf16, double-buffered
static constexpr int LDA_U = 260;                 // uints per A row (1040 B)
static constexpr int LDB_U = 68;                  // uints per B row (272 B)
static constexpr int SA_BYTES = 128 * LDA_U * 4;  // 133120
static constexpr int SB_BYTES = 128 * LDB_U * 4;  // 34816
static constexpr int SMEM_GEMM = SA_BYTES + 2 * SB_BYTES; // 202752

#if defined(__gfx1250__) && __has_builtin(__builtin_amdgcn_global_load_async_to_lds_b128) && \
    __has_builtin(__builtin_amdgcn_s_wait_asynccnt)
#define HAVE_ASYNC 1
#else
#define HAVE_ASYNC 0
#endif

#define AS_GLOBAL __attribute__((address_space(1)))
#define AS_LDS    __attribute__((address_space(3)))

#if HAVE_ASYNC
#define WAIT_ASYNC(n) __builtin_amdgcn_s_wait_asynccnt(n)
#define ASYNC_CP16(gsrc, ldst) \
  __builtin_amdgcn_global_load_async_to_lds_b128((AS_GLOBAL vint4*)(gsrc), (AS_LDS vint4*)(ldst), 0, 0)
#else
#define WAIT_ASYNC(n) ((void)0)
#endif

union FragAB { v16bf bf; uint4 q[2]; };

__device__ __forceinline__ unsigned short f2bf(float x) {
  unsigned u = __float_as_uint(x);
  u += 0x7FFFu + ((u >> 16) & 1u);          // round-to-nearest-even
  return (unsigned short)(u >> 16);
}
__device__ __forceinline__ unsigned pack2(float a, float b) {
  return (unsigned)f2bf(a) | ((unsigned)f2bf(b) << 16);
}

// ---- staging helpers (256-thread workgroups) ----
// A from f32 (with optional gather), converted to bf16 in-register, b128 global loads
__device__ __forceinline__ void stage_A_f32(unsigned* sA, const float* src,
                                            const int* gather, int rowBase, int tid) {
  for (int i = tid; i < 128 * 128; i += 256) {
    int r = i >> 7, c = i & 127;
    int gr = rowBase + r;
    if (gather) gr = gather[gr];
    const float4 v = ((const float4*)(src + (size_t)gr * H))[c];
    uint2 w;
    w.x = pack2(v.x, v.y);
    w.y = pack2(v.z, v.w);
    ((uint2*)(sA + r * LDA_U))[c] = w;
  }
}

// A tile from bf16 source (row stride 512 bf16 = 1024 B): 32 x b128 per thread
__device__ __forceinline__ void issue_A_bf16(char* sA, const unsigned* src_u,
                                             int rowBase, int tid) {
#if HAVE_ASYNC
#pragma unroll
  for (int j = 0; j < 32; ++j) {
    int i = tid + j * 256;
    int r = i >> 6, c = i & 63;
    const char* g = (const char*)src_u + (size_t)(rowBase + r) * 1024 + c * 16;
    char* l = sA + r * 1040 + c * 16;
    ASYNC_CP16(g, l);
  }
#else
  for (int i = tid; i < 128 * 64; i += 256) {
    int r = i >> 6, c = i & 63;
    ((uint4*)(sA + r * 1040))[c] =
        ((const uint4*)((const char*)src_u + (size_t)(rowBase + r) * 1024))[c];
  }
#endif
}

// B chunk: rows [nBase,nBase+128) x K [kc*128,+128) from bf16 source (row stride 1024 B)
// 8 x b128 per thread
__device__ __forceinline__ void issue_B(char* sBbuf, const unsigned* src_u,
                                        int nBase, int kc, int tid) {
#if HAVE_ASYNC
#pragma unroll
  for (int j = 0; j < 8; ++j) {
    int i = tid + j * 256;
    int n = i >> 4, c = i & 15;
    const char* g = (const char*)src_u + (size_t)(nBase + n) * 1024 + kc * 256 + c * 16;
    char* l = sBbuf + n * 272 + c * 16;
    ASYNC_CP16(g, l);
  }
#else
  for (int i = tid; i < 128 * 16; i += 256) {
    int n = i >> 4, c = i & 15;
    ((uint4*)(sBbuf + n * 272))[c] =
        ((const uint4*)((const char*)src_u + (size_t)(nBase + n) * 1024 + kc * 256))[c];
  }
#endif
}

// ---- fragment loads (wave32, layouts per ISA 7.12.2) ----
__device__ __forceinline__ v16bf load_a(const char* sA, int strip, int lane, int kb) {
  int row = strip + (lane & 15);
  int g   = lane >> 4;
  const char* p = sA + row * (LDA_U * 4) + kb * 2;
  FragAB f;
  f.q[0] = *(const uint4*)(p + g * 16);        // K = kb + 8g .. +8
  f.q[1] = *(const uint4*)(p + 32 + g * 16);   // K = kb + 16 + 8g .. +8
  return f.bf;
}
__device__ __forceinline__ v16bf load_b(const char* sB, int nt, int lane, int ks) {
  int n = nt * 16 + (lane & 15);
  int g = lane >> 4;
  const char* p = sB + n * (LDB_U * 4) + ks * 2;  // K = ks + 8g .. +8, ks + 16 + 8g .. +8
  FragAB f;
  f.q[0] = *(const uint4*)(p + g * 16);
  f.q[1] = *(const uint4*)(p + 32 + g * 16);
  return f.bf;
}

// 16x128-strip x 128-K-chunk of WMMAs into acc[8].
// Software-pipelined: next B (and next A across ks) loaded before the WMMA
// consuming the current fragments, so ds_load latency overlaps matrix work.
__device__ __forceinline__ void mma_block(v8f acc[8], const char* sA, const char* sB,
                                          int strip, int lane, int kbase) {
  v16bf a = load_a(sA, strip, lane, kbase);
  v16bf b = load_b(sB, 0, lane, 0);
#pragma unroll
  for (int ks = 0; ks < 128; ks += 32) {
    v16bf anext = a;
    if (ks < 96) anext = load_a(sA, strip, lane, kbase + ks + 32);
#pragma unroll
    for (int nt = 0; nt < 8; ++nt) {
      v16bf bnext = b;
      if (nt < 7)       bnext = load_b(sB, nt + 1, lane, ks);
      else if (ks < 96) bnext = load_b(sB, 0, lane, ks + 32);
      acc[nt] = __builtin_amdgcn_wmma_f32_16x16x32_bf16(
          false, a, false, b, (short)0, acc[nt], false, false);
      b = bnext;
    }
    a = anext;
  }
}

// ---------------- kernel 1: Q partial column sums ----------------
__global__ void k_colsum(const float* __restrict__ clause, float* __restrict__ Qpart) {
  int tid = threadIdx.x;                 // 512
  int b   = blockIdx.x;                  // 128 blocks x 512 rows
  const float* base = clause + (size_t)b * 512 * H;
  float s = 0.f;
  for (int r = 0; r < 512; ++r) s += base[(size_t)r * H + tid];
  Qpart[b * H + tid] = s;
}

// ---------------- kernel 2: Q reduce + Q_t (+ fold var_K_b) ----------------
__global__ void k_finalizeQ(const float* __restrict__ Qpart, const float* __restrict__ vQw,
                            const float* __restrict__ vQb, const float* __restrict__ vKb,
                            float* __restrict__ QtB) {
  __shared__ float sQ[H];
  int tid = threadIdx.x;                 // 512
  float s = 0.f;
  for (int b = 0; b < 128; ++b) s += Qpart[b * H + tid];
  sQ[tid] = s;
  __syncthreads();
  float acc = vQb[tid];
  const float* wr = vQw + (size_t)tid * H;
  for (int h = 0; h < H; ++h) acc += sQ[h] * wr[h];
  QtB[tid] = acc + vKb[tid];
}

// ---------------- kernel 3: f32 -> bf16 weight convert ----------------
__global__ void k_convert(const float* __restrict__ src, unsigned short* __restrict__ dst, int n) {
  int i = blockIdx.x * blockDim.x + threadIdx.x;
  if (i < n) dst[i] = f2bf(src[i]);
}

// ---------------- kernel 4: u = tanh(K@Wk^T + QtB) @ attn_w + attn_b ----------------
__global__ void __launch_bounds__(256)
k_gemm_u(const float* __restrict__ litK, const unsigned* __restrict__ Wk_u,
         const float* __restrict__ QtB, const float* __restrict__ attn_w,
         const float* __restrict__ attn_b, float* __restrict__ u) {
  extern __shared__ char smem[];
  unsigned* sA = (unsigned*)smem;
  char* sB0 = smem + SA_BYTES;
  char* sB1 = sB0 + SB_BYTES;
  int tid = threadIdx.x, lane = tid & 31;
  int strip = (tid >> 5) * 16;
  int rowBase = blockIdx.x * 128;
  int g = lane >> 4;

  stage_A_f32(sA, litK, nullptr, rowBase, tid);
  __syncthreads();

  float uacc[8] = {0.f, 0.f, 0.f, 0.f, 0.f, 0.f, 0.f, 0.f};
  for (int nb = 0; nb < 4; ++nb) {
    v8f acc[8];
#pragma unroll
    for (int nt = 0; nt < 8; ++nt) acc[nt] = (v8f){0.f,0.f,0.f,0.f,0.f,0.f,0.f,0.f};
    issue_B(sB0, Wk_u, nb * 128, 0, tid);
    for (int kc = 0; kc < 4; ++kc) {
      char* cur = (kc & 1) ? sB1 : sB0;
      if (kc < 3) { issue_B((kc & 1) ? sB0 : sB1, Wk_u, nb * 128, kc + 1, tid); WAIT_ASYNC(8); }
      else        { WAIT_ASYNC(0); }
      __syncthreads();
      mma_block(acc, (const char*)sA, cur, strip, lane, kc * 128);
      __syncthreads();
    }
#pragma unroll
    for (int nt = 0; nt < 8; ++nt) {
      int col = nb * 128 + nt * 16 + (lane & 15);
      float qt = QtB[col];
      float aw = attn_w[col];
#pragma unroll
      for (int r = 0; r < 8; ++r) uacc[r] += tanhf(acc[nt][r] + qt) * aw;
    }
  }
  float ab = attn_b[0];
#pragma unroll
  for (int r = 0; r < 8; ++r) {
    float v = uacc[r];
    v += __shfl_xor(v, 1, 16);
    v += __shfl_xor(v, 2, 16);
    v += __shfl_xor(v, 4, 16);
    v += __shfl_xor(v, 8, 16);
    if ((lane & 15) == 0) u[rowBase + strip + r + 8 * g] = v + ab;
  }
}

// ---------------- kernel 5: var_logp / var_idx over u ----------------
__global__ void k_ustats(const float* __restrict__ u, float* __restrict__ stats) {
  __shared__ float sm[512];
  __shared__ int   si[512];
  __shared__ float ss[512];
  int tid = threadIdx.x;
  float m = -__builtin_inff();
  int bi = 0;
  for (int i = tid; i < NVAR; i += 512) {
    float v = u[i];
    if (v > m) { m = v; bi = i; } else if (v == m && i < bi) bi = i;
  }
  sm[tid] = m; si[tid] = bi;
  __syncthreads();
  for (int s = 256; s > 0; s >>= 1) {
    if (tid < s) {
      if (sm[tid + s] > sm[tid] || (sm[tid + s] == sm[tid] && si[tid + s] < si[tid])) {
        sm[tid] = sm[tid + s]; si[tid] = si[tid + s];
      }
    }
    __syncthreads();
  }
  float mu = sm[0];
  float se = 0.f;
  for (int i = tid; i < NVAR; i += 512) se += __expf(u[i] - mu);
  ss[tid] = se;
  __syncthreads();
  for (int s = 256; s > 0; s >>= 1) {
    if (tid < s) ss[tid] += ss[tid + s];
    __syncthreads();
  }
  if (tid == 0) { stats[0] = -logf(ss[0]); ((int*)stats)[1] = si[0]; }
}

// ---------------- kernel 6: gather + project q,k (bf16 out) ----------------
__global__ void __launch_bounds__(256)
k_project(const float* __restrict__ clause, const int* __restrict__ pos_idx,
          const int* __restrict__ neg_idx, const unsigned* __restrict__ Wq_u,
          const unsigned* __restrict__ Wk_u, const float* __restrict__ bq,
          const float* __restrict__ bk, unsigned short* __restrict__ qb,
          unsigned short* __restrict__ kb) {
  extern __shared__ char smem[];
  unsigned* sA = (unsigned*)smem;
  char* sB0 = smem + SA_BYTES;
  char* sB1 = sB0 + SB_BYTES;
  bool isQ = (blockIdx.y == 0);
  const int* gather = isQ ? pos_idx : neg_idx;
  const unsigned* Wu = isQ ? Wq_u : Wk_u;
  const float* bias = isQ ? bq : bk;
  unsigned short* out = isQ ? qb : kb;

  int tid = threadIdx.x, lane = tid & 31;
  int strip = (tid >> 5) * 16;
  int rowBase = blockIdx.x * 128;
  int g = lane >> 4;

  stage_A_f32(sA, clause, gather, rowBase, tid);
  __syncthreads();

  for (int nb = 0; nb < 4; ++nb) {
    v8f acc[8];
#pragma unroll
    for (int nt = 0; nt < 8; ++nt) acc[nt] = (v8f){0.f,0.f,0.f,0.f,0.f,0.f,0.f,0.f};
    issue_B(sB0, Wu, nb * 128, 0, tid);
    for (int kc = 0; kc < 4; ++kc) {
      char* cur = (kc & 1) ? sB1 : sB0;
      if (kc < 3) { issue_B((kc & 1) ? sB0 : sB1, Wu, nb * 128, kc + 1, tid); WAIT_ASYNC(8); }
      else        { WAIT_ASYNC(0); }
      __syncthreads();
      mma_block(acc, (const char*)sA, cur, strip, lane, kc * 128);
      __syncthreads();
    }
#pragma unroll
    for (int nt = 0; nt < 8; ++nt) {
      int col = nb * 128 + nt * 16 + (lane & 15);
      float bv = bias[col];
#pragma unroll
      for (int r = 0; r < 8; ++r) {
        int row = rowBase + strip + r + 8 * g;
        out[(size_t)row * H + col] = f2bf(acc[nt][r] + bv);
      }
    }
  }
}

// streaming-logsumexp merge of (m, s, idx) triples
__device__ __forceinline__ void lse_merge(float& m, float& ssum, int& bidx,
                                          float om, float os, int oi) {
  if (om > m)       { ssum = ssum * __expf(m - om) + os; m = om; bidx = oi; }
  else if (om == m) { ssum += os; bidx = min(bidx, oi); }
  else              { ssum += os * __expf(om - m); }
}

// ---------------- kernel 7: scores GEMM + mask + per-WG (max,sumexp,argmax) ----------------
__global__ void __launch_bounds__(256)
k_scores(const unsigned* __restrict__ qb_u, const unsigned* __restrict__ kb_u,
         const unsigned char* __restrict__ keep, const unsigned char* __restrict__ taken,
         float* __restrict__ partM, float* __restrict__ partS, int* __restrict__ partI) {
  extern __shared__ char smem[];
  unsigned* sA = (unsigned*)smem;
  char* sB0 = smem + SA_BYTES;
  char* sB1 = sB0 + SB_BYTES;
  int tid = threadIdx.x, lane = tid & 31, w = tid >> 5;
  int strip = w * 16;
  int rowBase = blockIdx.x * 128, colBase = blockIdx.y * 128;
  int g = lane >> 4;

  issue_A_bf16((char*)sA, qb_u, rowBase, tid);
  issue_B(sB0, kb_u, colBase, 0, tid);

  v8f acc[8];
#pragma unroll
  for (int nt = 0; nt < 8; ++nt) acc[nt] = (v8f){0.f,0.f,0.f,0.f,0.f,0.f,0.f,0.f};
  for (int kc = 0; kc < 4; ++kc) {
    char* cur = (kc & 1) ? sB1 : sB0;
    if (kc < 3) { issue_B((kc & 1) ? sB0 : sB1, kb_u, colBase, kc + 1, tid); WAIT_ASYNC(8); }
    else        { WAIT_ASYNC(0); }
    __syncthreads();
    mma_block(acc, (const char*)sA, cur, strip, lane, kc * 128);
    __syncthreads();
  }

  float m = -__builtin_inff(), ssum = 0.f;
  int bidx = 0;
#pragma unroll
  for (int nt = 0; nt < 8; ++nt) {
    int col = colBase + nt * 16 + (lane & 15);
#pragma unroll
    for (int r = 0; r < 8; ++r) {
      int row = rowBase + strip + r + 8 * g;
      size_t fi = (size_t)row * NM + col;
      bool valid = keep[fi] && !taken[fi];
      float s = valid ? acc[nt][r] * RSQRTH : NEGV;
      if (s > m)       { ssum = ssum * __expf(m - s) + 1.f; m = s; bidx = (int)fi; }
      else if (s == m) { ssum += 1.f; if ((int)fi < bidx) bidx = (int)fi; }
      else             { ssum += __expf(s - m); }
    }
  }
  for (int d = 16; d >= 1; d >>= 1) {
    float om = __shfl_xor(m, d, 32);
    float os = __shfl_xor(ssum, d, 32);
    int   oi = __shfl_xor(bidx, d, 32);
    lse_merge(m, ssum, bidx, om, os, oi);
  }
  __syncthreads();                         // repurpose LDS for cross-wave reduce
  float* rm = (float*)smem;
  float* rs = rm + 8;
  int*   ri = (int*)(rm + 16);
  if (lane == 0) { rm[w] = m; rs[w] = ssum; ri[w] = bidx; }
  __syncthreads();
  if (tid == 0) {
    float M = rm[0], S = rs[0];
    int I = ri[0];
    for (int i = 1; i < 8; ++i) lse_merge(M, S, I, rm[i], rs[i], ri[i]);
    int wg = blockIdx.y * gridDim.x + blockIdx.x;
    partM[wg] = M; partS[wg] = S; partI[wg] = I;
  }
}

// ---------------- kernel 8: final merge + outputs ----------------
__global__ void k_final(const float* __restrict__ partM, const float* __restrict__ partS,
                        const int* __restrict__ partI, const float* __restrict__ stats,
                        const int* __restrict__ pos_idx, const int* __restrict__ neg_idx,
                        float* __restrict__ out) {
  if (threadIdx.x == 0 && blockIdx.x == 0) {
    float M = partM[0], S = partS[0];
    int I = partI[0];
    for (int i = 1; i < 1024; ++i) lse_merge(M, S, I, partM[i], partS[i], partI[i]);
    float C_logp   = -logf(S);               // max(logp) = M - (M + log S)
    float var_logp = stats[0];
    int var_idx    = ((const int*)stats)[1];
    int ci = I / NM, cj = I % NM;
    out[0] = C_logp + var_logp;
    out[1] = (float)pos_idx[ci];
    out[2] = (float)neg_idx[cj];
    out[3] = (float)var_idx;
  }
}

extern "C" void kernel_launch(void* const* d_in, const int* in_sizes, int n_in,
                              void* d_out, int out_size, void* d_ws, size_t ws_size,
                              hipStream_t stream) {
  (void)in_sizes; (void)n_in; (void)out_size; (void)ws_size;
  const float* literal_emb = (const float*)d_in[0];
  const float* clause_emb  = (const float*)d_in[1];
  const int*   pos_idx     = (const int*)d_in[2];
  const int*   neg_idx     = (const int*)d_in[3];
  const unsigned char* keep  = (const unsigned char*)d_in[4];
  const unsigned char* taken = (const unsigned char*)d_in[5];
  const float* var_K_w = (const float*)d_in[6];
  const float* var_K_b = (const float*)d_in[7];
  const float* var_Q_w = (const float*)d_in[8];
  const float* var_Q_b = (const float*)d_in[9];
  const float* var_attn_w = (const float*)d_in[10];
  const float* var_attn_b = (const float*)d_in[11];
  const float* W_Q_w = (const float*)d_in[12];
  const float* W_Q_b = (const float*)d_in[13];
  const float* W_K_w = (const float*)d_in[14];
  const float* W_K_b = (const float*)d_in[15];

  char* ws = (char*)d_ws;
  size_t o = 0;
  auto take = [&](size_t b) { char* p = ws + o; o += (b + 255) & ~(size_t)255; return p; };
  float* Qpart          = (float*)take((size_t)128 * H * 4);
  float* QtB            = (float*)take((size_t)H * 4);
  float* uStats         = (float*)take(256);
  float* u              = (float*)take((size_t)NVAR * 4);
  unsigned short* Wk_bf = (unsigned short*)take((size_t)H * H * 2);
  unsigned short* Wq_bf = (unsigned short*)take((size_t)H * H * 2);
  unsigned short* Wk2_bf= (unsigned short*)take((size_t)H * H * 2);
  unsigned short* qb    = (unsigned short*)take((size_t)NP * H * 2);
  unsigned short* kb    = (unsigned short*)take((size_t)NM * H * 2);
  float* partM          = (float*)take(1024 * 4);
  float* partS          = (float*)take(1024 * 4);
  int*   partI          = (int*)take(1024 * 4);

  k_colsum<<<128, 512, 0, stream>>>(clause_emb, Qpart);
  k_finalizeQ<<<1, 512, 0, stream>>>(Qpart, var_Q_w, var_Q_b, var_K_b, QtB);
  k_convert<<<(H * H + 255) / 256, 256, 0, stream>>>(var_K_w, Wk_bf, H * H);
  k_convert<<<(H * H + 255) / 256, 256, 0, stream>>>(W_Q_w, Wq_bf, H * H);
  k_convert<<<(H * H + 255) / 256, 256, 0, stream>>>(W_K_w, Wk2_bf, H * H);

  k_gemm_u<<<NVAR / 128, 256, SMEM_GEMM, stream>>>(
      literal_emb, (const unsigned*)Wk_bf, QtB, var_attn_w, var_attn_b, u);
  k_ustats<<<1, 512, 0, stream>>>(u, uStats);

  k_project<<<dim3(NP / 128, 2), 256, SMEM_GEMM, stream>>>(
      clause_emb, pos_idx, neg_idx, (const unsigned*)Wq_bf, (const unsigned*)Wk2_bf,
      W_Q_b, W_K_b, qb, kb);

  k_scores<<<dim3(NP / 128, NM / 128), 256, SMEM_GEMM, stream>>>(
      (const unsigned*)qb, (const unsigned*)kb, keep, taken, partM, partS, partI);

  k_final<<<1, 32, 0, stream>>>(partM, partS, partI, uStats, pos_idx, neg_idx, (float*)d_out);
}
